// GPT_5162550690660
// MI455X (gfx1250) — matrix operand order repaired and checked by reading, not compiled, not run on hardware
//
#include <hip/hip_runtime.h>
#include <hip/hip_bf16.h>

// ---------------- constants ----------------
#define LNUM 4
#define HNUM 12
#define EDIM 768
#define TLEN 1024
#define BNUM 4
#define VNUM 50257
#define DHEAD 64
#define E3 (3*EDIM)
#define E4 (4*EDIM)
#define ROWS (BNUM*TLEN)   // 4096

typedef __attribute__((ext_vector_type(16))) __bf16 v16bf;
typedef __attribute__((ext_vector_type(8)))  float  v8f;

// ---------------- helpers ----------------
// Hardware packed f32->bf16 conversion (RNE).
__device__ __forceinline__ unsigned pk2(float a, float b) {
    unsigned d;
    asm("v_cvt_pk_bf16_f32 %0, %1, %2" : "=v"(d) : "v"(a), "v"(b));
    return d;
}
__device__ __forceinline__ unsigned short f2bf(float x) {
    return (unsigned short)pk2(x, 0.0f);
}

// CDNA5 async global->LDS copy (16B per lane), tracked by ASYNCcnt.
__device__ __forceinline__ void async_copy16(unsigned ldsAddr, const void* gptr) {
    asm volatile("global_load_async_to_lds_b128 %0, %1, off"
                 :: "v"(ldsAddr), "v"((unsigned long long)(size_t)gptr)
                 : "memory");
}
__device__ __forceinline__ void wait_async0() {
    asm volatile("s_wait_asynccnt 0x0" ::: "memory");
}

// A-operand fragment (16 rows x 32 K): lane = row (0-15), khalf = lane>>4.
// word layout per ISA 7.12.2: j<4 -> word j, j>=4 -> word j+4, +4*khalf.
__device__ __forceinline__ v16bf frag_a(const unsigned* s, int rowWordBase, int khalf) {
    union { v16bf v; unsigned u[8]; } f;
    const unsigned* p = s + rowWordBase + 4 * khalf;
    f.u[0] = p[0]; f.u[1] = p[1]; f.u[2] = p[2]; f.u[3] = p[3];
    f.u[4] = p[8]; f.u[5] = p[9]; f.u[6] = p[10]; f.u[7] = p[11];
    return f.v;
}
// B-operand fragment: lane = column (0-15), same word layout as A:
// j<4 -> word j, j>=4 -> word j+4, +4*khalf.
__device__ __forceinline__ v16bf frag_b(const unsigned* s, int rowWordBase, int khalf) {
    union { v16bf v; unsigned u[8]; } f;
    const unsigned* p = s + rowWordBase + 4 * khalf;
#pragma unroll
    for (int j = 0; j < 8; ++j) f.u[j] = p[j < 4 ? j : j + 4];
    return f.v;
}

union Facc { v8f v; float f[8]; };

// ---------------- embedding ----------------
__global__ __launch_bounds__(256) void embed_kernel(const int* __restrict__ idx,
                                                    const float* __restrict__ wte,
                                                    const float* __restrict__ wpe,
                                                    float* __restrict__ x) {
    int row = blockIdx.x;                 // b*T + t
    int t = row & (TLEN - 1);
    int tok = idx[row];
    const float* we = wte + (size_t)tok * EDIM;
    const float* pe = wpe + (size_t)t * EDIM;
    float* xr = x + (size_t)row * EDIM;
    for (int e = threadIdx.x; e < EDIM; e += 256) xr[e] = we[e] + pe[e];
}

// ---------------- layernorm -> bf16 ----------------
__global__ __launch_bounds__(256) void layernorm_kernel(const float* __restrict__ x,
                                                        const float* __restrict__ g,
                                                        const float* __restrict__ bta,
                                                        unsigned short* __restrict__ out) {
    __shared__ float red[256];
    int row = blockIdx.x;
    int tid = threadIdx.x;
    const float* xr = x + (size_t)row * EDIM;
    float v0 = xr[tid], v1 = xr[tid + 256], v2 = xr[tid + 512];
    red[tid] = v0 + v1 + v2;
    __syncthreads();
    for (int s = 128; s > 0; s >>= 1) { if (tid < s) red[tid] += red[tid + s]; __syncthreads(); }
    float mean = red[0] * (1.0f / EDIM);
    __syncthreads();
    float d0 = v0 - mean, d1 = v1 - mean, d2 = v2 - mean;
    red[tid] = d0 * d0 + d1 * d1 + d2 * d2;
    __syncthreads();
    for (int s = 128; s > 0; s >>= 1) { if (tid < s) red[tid] += red[tid + s]; __syncthreads(); }
    float rstd = rsqrtf(red[0] * (1.0f / EDIM) + 1e-5f);
    unsigned short* o = out + (size_t)row * EDIM;
    o[tid]       = f2bf(d0 * rstd * g[tid]       + bta[tid]);
    o[tid + 256] = f2bf(d1 * rstd * g[tid + 256] + bta[tid + 256]);
    o[tid + 512] = f2bf(d2 * rstd * g[tid + 512] + bta[tid + 512]);
}

// ---------------- generic WMMA GEMM: C = A_bf16 @ W_f32^T (+bias)(+residual) ----------------
// A: [M,K] bf16 row-major. W: [N,K] f32 row-major. C: [M,N] f32.
// Block: 256 threads (8 waves), tile 128x128, BK=32, double-buffered LDS.
// A staged via GLOBAL_LOAD_ASYNC_TO_LDS_B128; W staged via VGPRs (needs f32->bf16).
// Waves 2(M)x4(N); each wave owns 64(M)x32(N) = eight 16x16 accumulators.
__global__ __launch_bounds__(256) void gemm_bf16(const unsigned short* __restrict__ A,
                                                 const float* __restrict__ W,
                                                 const float* __restrict__ bias,
                                                 const float* __restrict__ residual,
                                                 float* __restrict__ C,
                                                 int M, int N, int K) {
    __shared__ unsigned short As[2][128 * 32];
    __shared__ unsigned short Ws[2][128 * 32];

    int n0 = blockIdx.x * 128, m0 = blockIdx.y * 128;
    int tid = threadIdx.x, lane = tid & 31, wvid = tid >> 5;
    int wm = wvid >> 2, wn = wvid & 3;
    int khalf = lane >> 4, l16 = lane & 15;
    int sr = tid >> 1, sc = (tid & 1) * 16;        // staging: row (0..127), col16

    const unsigned short* ap = A + (size_t)(m0 + sr) * K + sc;
    const float* wp0 = W + (size_t)(n0 + sr) * K + sc;
    const bool wvalid = (n0 + sr) < N;

    // LDS byte address of this thread's A staging slot (low 32 bits of flat addr).
    const unsigned ldsA = (unsigned)(unsigned long long)(size_t)&As[0][sr * 32 + sc];

    Facc acc[4][2];
#pragma unroll
    for (int t = 0; t < 4; ++t)
#pragma unroll
        for (int u = 0; u < 2; ++u)
#pragma unroll
            for (int i = 0; i < 8; ++i) acc[t][u].f[i] = 0.f;

    auto asyncA = [&](int kk, int pp) {
        unsigned d = ldsA + (unsigned)pp * (128 * 32 * 2);
        const unsigned short* s = ap + kk;
        async_copy16(d, s);
        async_copy16(d + 16, s + 8);
    };
    auto loadW = [&](int kk, uint4& w0, uint4& w1) {
        w0.x = w0.y = w0.z = w0.w = 0u;
        w1 = w0;
        if (wvalid) {
            const float* wp = wp0 + kk;
            float4 f0 = *(const float4*)(wp);
            float4 f1 = *(const float4*)(wp + 4);
            float4 f2 = *(const float4*)(wp + 8);
            float4 f3 = *(const float4*)(wp + 12);
            w0.x = pk2(f0.x, f0.y); w0.y = pk2(f0.z, f0.w);
            w0.z = pk2(f1.x, f1.y); w0.w = pk2(f1.z, f1.w);
            w1.x = pk2(f2.x, f2.y); w1.y = pk2(f2.z, f2.w);
            w1.z = pk2(f3.x, f3.y); w1.w = pk2(f3.z, f3.w);
        }
    };
    auto storeW = [&](int pp, uint4 w0, uint4 w1) {
        uint4* dw = (uint4*)(&Ws[pp][sr * 32 + sc]);
        dw[0] = w0; dw[1] = w1;
    };

    // prologue: stage k-tile 0 into buffer 0
    {
        uint4 w0, w1;
        asyncA(0, 0);
        loadW(0, w0, w1);
        storeW(0, w0, w1);
        wait_async0();
    }

    int p = 0;
    for (int k0 = 0; k0 < K; k0 += 32, p ^= 1) {
        bool hasNext = (k0 + 32 < K);
        uint4 nw0, nw1;
        if (hasNext) loadW(k0 + 32, nw0, nw1);

        __syncthreads();   // buffer p published to all waves; buffer p^1 free

        if (hasNext) asyncA(k0 + 32, p ^ 1);   // overlaps with the WMMA chain below

        const unsigned* As32 = (const unsigned*)As[p];
        const unsigned* Ws32 = (const unsigned*)Ws[p];
        v16bf bf0 = frag_b(Ws32, (wn * 32 + l16) * 16, khalf);
        v16bf bf1 = frag_b(Ws32, (wn * 32 + 16 + l16) * 16, khalf);
#pragma unroll
        for (int t = 0; t < 4; ++t) {
            v16bf af = frag_a(As32, (wm * 64 + t * 16 + l16) * 16, khalf);
            acc[t][0].v = __builtin_amdgcn_wmma_f32_16x16x32_bf16(false, af, false, bf0,
                                                                  (short)0, acc[t][0].v, false, false);
            acc[t][1].v = __builtin_amdgcn_wmma_f32_16x16x32_bf16(false, af, false, bf1,
                                                                  (short)0, acc[t][1].v, false, false);
        }

        if (hasNext) {
            storeW(p ^ 1, nw0, nw1);
            wait_async0();   // async A(k+1) complete before next barrier
        }
    }

#pragma unroll
    for (int u = 0; u < 2; ++u) {
        int n = n0 + wn * 32 + u * 16 + l16;
        if (n < N) {
            float bv = bias ? bias[n] : 0.f;
#pragma unroll
            for (int t = 0; t < 4; ++t)
#pragma unroll
                for (int r = 0; r < 8; ++r) {
                    int m = m0 + wm * 64 + t * 16 + r + 8 * khalf;
                    float val = acc[t][u].f[r] + bv;
                    if (residual) val += residual[(size_t)m * N + n];
                    C[(size_t)m * N + n] = val;
                }
        }
    }
}

// ---------------- flash attention (causal) ----------------
// qkv: [ROWS, 3E] f32.  y: [ROWS, E] bf16.  grid(T/64, H, B), block 128 (4 waves).
__global__ __launch_bounds__(128) void flash_attn(const float* __restrict__ qkv,
                                                  unsigned short* __restrict__ y) {
    __shared__ unsigned short Qs[64 * 64];
    __shared__ unsigned short Ks[64 * 64];
    __shared__ unsigned short Vt[64 * 64];      // transposed: [dim][key]
    __shared__ unsigned short Ps[4 * 16 * 64];  // per-wave P tiles

    int qt = blockIdx.x, h = blockIdx.y, b = blockIdx.z;
    int qbase = qt * 64;
    int tid = threadIdx.x, lane = tid & 31, wvid = tid >> 5;
    int khalf = lane >> 4, l16 = lane & 15;
    const float scale = 0.125f; // 1/sqrt(64)

    // stage Q (scaled): 2 threads per row, 32 dims each
    {
        int rr = tid >> 1, ch = (tid & 1) * 32;
        const float* src = qkv + (size_t)(b * TLEN + qbase + rr) * E3 + h * DHEAD + ch;
        unsigned* dst = (unsigned*)(Qs + rr * 64 + ch);
#pragma unroll
        for (int i = 0; i < 8; ++i) {
            float4 f = *(const float4*)(src + i * 4);
            dst[i * 2]     = pk2(f.x * scale, f.y * scale);
            dst[i * 2 + 1] = pk2(f.z * scale, f.w * scale);
        }
    }
    __syncthreads();

    v16bf qf0 = frag_a((unsigned*)Qs, (wvid * 16 + l16) * 32 + 0,  khalf);
    v16bf qf1 = frag_a((unsigned*)Qs, (wvid * 16 + l16) * 32 + 16, khalf);

    float mrow[8], lrow[8];
    Facc O[4];
#pragma unroll
    for (int r = 0; r < 8; ++r) { mrow[r] = -1e30f; lrow[r] = 0.f; }
#pragma unroll
    for (int c = 0; c < 4; ++c)
#pragma unroll
        for (int r = 0; r < 8; ++r) O[c].f[r] = 0.f;

    unsigned* Ps32w = (unsigned*)(Ps + wvid * 16 * 64);

    for (int j = 0; j <= qt; ++j) {
        int kb = j * 64;
        __syncthreads();
        // stage K [key][dim] and V transposed [dim][key]
        {
            int rr = tid >> 1, ch = (tid & 1) * 32;
            const float* ks = qkv + (size_t)(b * TLEN + kb + rr) * E3 + EDIM + h * DHEAD + ch;
            unsigned* dst = (unsigned*)(Ks + rr * 64 + ch);
#pragma unroll
            for (int i = 0; i < 8; ++i) {
                float4 f = *(const float4*)(ks + i * 4);
                dst[i * 2]     = pk2(f.x, f.y);
                dst[i * 2 + 1] = pk2(f.z, f.w);
            }
            const float* vs = qkv + (size_t)(b * TLEN + kb + rr) * E3 + 2 * EDIM + h * DHEAD + ch;
#pragma unroll
            for (int i = 0; i < 8; ++i) {
                float4 f = *(const float4*)(vs + i * 4);
                int d = ch + i * 4;
                unsigned p01 = pk2(f.x, f.y), p23 = pk2(f.z, f.w);
                Vt[(d + 0) * 64 + rr] = (unsigned short)p01;
                Vt[(d + 1) * 64 + rr] = (unsigned short)(p01 >> 16);
                Vt[(d + 2) * 64 + rr] = (unsigned short)p23;
                Vt[(d + 3) * 64 + rr] = (unsigned short)(p23 >> 16);
            }
        }
        __syncthreads();

        // S = Q @ K^T  (4 key tiles of 16)
        Facc S[4];
#pragma unroll
        for (int c = 0; c < 4; ++c) {
#pragma unroll
            for (int r = 0; r < 8; ++r) S[c].f[r] = 0.f;
            v16bf b0 = frag_b((unsigned*)Ks, (c * 16 + l16) * 32 + 0,  khalf);
            v16bf b1 = frag_b((unsigned*)Ks, (c * 16 + l16) * 32 + 16, khalf);
            S[c].v = __builtin_amdgcn_wmma_f32_16x16x32_bf16(false, qf0, false, b0, (short)0, S[c].v, false, false);
            S[c].v = __builtin_amdgcn_wmma_f32_16x16x32_bf16(false, qf1, false, b1, (short)0, S[c].v, false, false);
        }

        // causal mask + online softmax
        float mloc[8];
#pragma unroll
        for (int r = 0; r < 8; ++r) mloc[r] = -1e30f;
#pragma unroll
        for (int c = 0; c < 4; ++c) {
            int kg = kb + c * 16 + l16;
#pragma unroll
            for (int r = 0; r < 8; ++r) {
                int qg = qbase + wvid * 16 + r + 8 * khalf;
                float s = (kg <= qg) ? S[c].f[r] : -1e30f;
                S[c].f[r] = s;
                mloc[r] = fmaxf(mloc[r], s);
            }
        }
#pragma unroll
        for (int r = 0; r < 8; ++r) {
            mloc[r] = fmaxf(mloc[r], __shfl_xor(mloc[r], 1, 32));
            mloc[r] = fmaxf(mloc[r], __shfl_xor(mloc[r], 2, 32));
            mloc[r] = fmaxf(mloc[r], __shfl_xor(mloc[r], 4, 32));
            mloc[r] = fmaxf(mloc[r], __shfl_xor(mloc[r], 8, 32));
        }
        float alpha[8], lsum[8];
#pragma unroll
        for (int r = 0; r < 8; ++r) {
            float mnew = fmaxf(mrow[r], mloc[r]);
            alpha[r] = __expf(mrow[r] - mnew);
            mrow[r] = mnew;
            lsum[r] = 0.f;
        }
#pragma unroll
        for (int c = 0; c < 4; ++c)
#pragma unroll
            for (int r = 0; r < 8; ++r) {
                float e = __expf(S[c].f[r] - mrow[r]);
                S[c].f[r] = e;
                lsum[r] += e;
            }
#pragma unroll
        for (int r = 0; r < 8; ++r) {
            lsum[r] += __shfl_xor(lsum[r], 1, 32);
            lsum[r] += __shfl_xor(lsum[r], 2, 32);
            lsum[r] += __shfl_xor(lsum[r], 4, 32);
            lsum[r] += __shfl_xor(lsum[r], 8, 32);
            lrow[r] = lrow[r] * alpha[r] + lsum[r];
        }
#pragma unroll
        for (int c = 0; c < 4; ++c)
#pragma unroll
            for (int r = 0; r < 8; ++r) O[c].f[r] *= alpha[r];

        // P -> bf16 in wave-private LDS (C-layout -> A-layout re-fragmentation)
        unsigned short* pw = Ps + wvid * 16 * 64;
#pragma unroll
        for (int c = 0; c < 4; ++c)
#pragma unroll
            for (int r = 0; r < 8; ++r)
                pw[(r + 8 * khalf) * 64 + c * 16 + l16] = f2bf(S[c].f[r]);
        // same-wave DS ordering guarantees store->load visibility

        v16bf pa0 = frag_a(Ps32w, l16 * 32 + 0,  khalf);
        v16bf pa1 = frag_a(Ps32w, l16 * 32 + 16, khalf);
#pragma unroll
        for (int c = 0; c < 4; ++c) {
            v16bf vb0 = frag_b((unsigned*)Vt, (c * 16 + l16) * 32 + 0,  khalf);
            v16bf vb1 = frag_b((unsigned*)Vt, (c * 16 + l16) * 32 + 16, khalf);
            O[c].v = __builtin_amdgcn_wmma_f32_16x16x32_bf16(false, pa0, false, vb0, (short)0, O[c].v, false, false);
            O[c].v = __builtin_amdgcn_wmma_f32_16x16x32_bf16(false, pa1, false, vb1, (short)0, O[c].v, false, false);
        }
    }

    // epilogue: O / l -> y (bf16)
#pragma unroll
    for (int c = 0; c < 4; ++c)
#pragma unroll
        for (int r = 0; r < 8; ++r) {
            int m = qbase + wvid * 16 + r + 8 * khalf;
            float o = O[c].f[r] / lrow[r];
            y[(size_t)(b * TLEN + m) * EDIM + h * DHEAD + c * 16 + l16] = f2bf(o);
        }
}

// ---------------- GELU -> bf16 (2 elements/thread, packed) ----------------
__global__ __launch_bounds__(256) void gelu_kernel(const float* __restrict__ in,
                                                   unsigned* __restrict__ out, int n2) {
    int i = blockIdx.x * 256 + threadIdx.x;
    if (i < n2) {
        float2 v = ((const float2*)in)[i];
        float u0 = 0.7978845608028654f * (v.x + 0.044715f * v.x * v.x * v.x);
        float u1 = 0.7978845608028654f * (v.y + 0.044715f * v.y * v.y * v.y);
        float g0 = 0.5f * v.x * (1.0f + tanhf(u0));
        float g1 = 0.5f * v.y * (1.0f + tanhf(u1));
        out[i] = pk2(g0, g1);
    }
}

// ---------------- cross-entropy ----------------
__global__ __launch_bounds__(256) void loss_row_kernel(const float* __restrict__ logits,
                                                       const int* __restrict__ targets,
                                                       float* __restrict__ rowloss) {
    __shared__ float red[256];
    int i = blockIdx.x;                // 0 .. B*(T-1)-1
    int b = i / (TLEN - 1), t = i % (TLEN - 1);
    size_t row = (size_t)(b * TLEN + t) * VNUM;
    int tid = threadIdx.x;
    float mx = -1e30f;
    for (int v = tid; v < VNUM; v += 256) mx = fmaxf(mx, logits[row + v]);
    red[tid] = mx;
    __syncthreads();
    for (int s = 128; s > 0; s >>= 1) { if (tid < s) red[tid] = fmaxf(red[tid], red[tid + s]); __syncthreads(); }
    float M = red[0];
    __syncthreads();
    float ssum = 0.f;
    for (int v = tid; v < VNUM; v += 256) ssum += __expf(logits[row + v] - M);
    red[tid] = ssum;
    __syncthreads();
    for (int s = 128; s > 0; s >>= 1) { if (tid < s) red[tid] += red[tid + s]; __syncthreads(); }
    if (tid == 0) {
        int lbl = targets[b * TLEN + t + 1];
        rowloss[i] = M + logf(red[0]) - logits[row + lbl];
    }
}

__global__ __launch_bounds__(256) void loss_reduce_kernel(const float* __restrict__ rowloss,
                                                          float* __restrict__ out) {
    __shared__ float red[256];
    int tid = threadIdx.x;
    const int n = BNUM * (TLEN - 1);
    float s = 0.f;
    for (int i = tid; i < n; i += 256) s += rowloss[i];
    red[tid] = s;
    __syncthreads();
    for (int st = 128; st > 0; st >>= 1) { if (tid < st) red[tid] += red[tid + st]; __syncthreads(); }
    if (tid == 0) out[0] = red[0] / (float)n;
}

// ---------------- host launcher ----------------
extern "C" void kernel_launch(void* const* d_in, const int* in_sizes, int n_in,
                              void* d_out, int out_size, void* d_ws, size_t ws_size,
                              hipStream_t stream) {
    const int*   idx     = (const int*)  d_in[0];
    const int*   targets = (const int*)  d_in[1];
    const float* wte     = (const float*)d_in[2];
    const float* wpe     = (const float*)d_in[3];
    const float* ln1_w   = (const float*)d_in[4];
    const float* ln1_b   = (const float*)d_in[5];
    const float* attn_w  = (const float*)d_in[6];
    const float* attn_b  = (const float*)d_in[7];
    const float* proj_w  = (const float*)d_in[8];
    const float* proj_b  = (const float*)d_in[9];
    const float* ln2_w   = (const float*)d_in[10];
    const float* ln2_b   = (const float*)d_in[11];
    const float* fc_w    = (const float*)d_in[12];
    const float* fc_b    = (const float*)d_in[13];
    const float* out_w   = (const float*)d_in[14];
    const float* out_b   = (const float*)d_in[15];
    const float* lnf_w   = (const float*)d_in[16];
    const float* lnf_b   = (const float*)d_in[17];

    float* logits = (float*)d_out;
    float* lossp  = logits + (size_t)ROWS * VNUM;

    // workspace carve-up (256B aligned)
    char* w = (char*)d_ws;
    size_t off = 0;
    auto carve = [&](size_t bytes) { void* p = w + off; off += (bytes + 255) & ~(size_t)255; return p; };
    float*          x       = (float*)         carve((size_t)ROWS * EDIM * 4);
    unsigned short* ln_bf   = (unsigned short*)carve((size_t)ROWS * EDIM * 2);
    unsigned short* y_bf    = (unsigned short*)carve((size_t)ROWS * EDIM * 2);
    float*          big     = (float*)         carve((size_t)ROWS * E4 * 4);   // qkv / fc_out
    unsigned short* gelu_bf = (unsigned short*)carve((size_t)ROWS * E4 * 2);
    float*          rowloss = (float*)         carve((size_t)BNUM * (TLEN - 1) * 4);
    (void)ws_size; (void)n_in; (void)in_sizes; (void)out_size;

    embed_kernel<<<ROWS, 256, 0, stream>>>(idx, wte, wpe, x);

    for (int l = 0; l < LNUM; ++l) {
        const float* aw = attn_w + (size_t)l * E3 * EDIM;
        const float* ab = attn_b + (size_t)l * E3;
        const float* pw = proj_w + (size_t)l * EDIM * EDIM;
        const float* pb = proj_b + (size_t)l * EDIM;
        const float* fw = fc_w   + (size_t)l * E4 * EDIM;
        const float* fb = fc_b   + (size_t)l * E4;
        const float* ow = out_w  + (size_t)l * EDIM * E4;
        const float* ob = out_b  + (size_t)l * EDIM;

        layernorm_kernel<<<ROWS, 256, 0, stream>>>(x, ln1_w + l * EDIM, ln1_b + l * EDIM, ln_bf);
        gemm_bf16<<<dim3(E3 / 128, ROWS / 128), 256, 0, stream>>>(ln_bf, aw, ab, nullptr, big,
                                                                  ROWS, E3, EDIM);
        flash_attn<<<dim3(TLEN / 64, HNUM, BNUM), 128, 0, stream>>>(big, y_bf);
        gemm_bf16<<<dim3(EDIM / 128, ROWS / 128), 256, 0, stream>>>(y_bf, pw, pb, x, x,
                                                                    ROWS, EDIM, EDIM);
        layernorm_kernel<<<ROWS, 256, 0, stream>>>(x, ln2_w + l * EDIM, ln2_b + l * EDIM, ln_bf);
        gemm_bf16<<<dim3(E4 / 128, ROWS / 128), 256, 0, stream>>>(ln_bf, fw, fb, nullptr, big,
                                                                  ROWS, E4, EDIM);
        int ng2 = ROWS * E4 / 2;
        gelu_kernel<<<(ng2 + 255) / 256, 256, 0, stream>>>(big, (unsigned*)gelu_bf, ng2);
        gemm_bf16<<<dim3(EDIM / 128, ROWS / 128), 256, 0, stream>>>(gelu_bf, ow, ob, x, x,
                                                                    ROWS, EDIM, E4);
    }

    layernorm_kernel<<<ROWS, 256, 0, stream>>>(x, lnf_w, lnf_b, ln_bf);
    gemm_bf16<<<dim3((VNUM + 127) / 128, ROWS / 128), 256, 0, stream>>>(ln_bf, wte, nullptr, nullptr,
                                                                        logits, ROWS, VNUM, EDIM);
    loss_row_kernel<<<BNUM * (TLEN - 1), 256, 0, stream>>>(logits, targets, rowloss);
    loss_reduce_kernel<<<1, 256, 0, stream>>>(rowloss, lossp);
}
